// GCNNe_close_to_ParticleNet_50465865728553
// MI455X (gfx1250) — matrix operand order repaired and checked by reading, not compiled, not run on hardware
//
#include <hip/hip_runtime.h>
#include <hip/hip_bf16.h>

typedef __attribute__((ext_vector_type(16))) _Float16 v16h;
typedef __attribute__((ext_vector_type(8)))  _Float16 v8h;
typedef __attribute__((ext_vector_type(8)))  float    v8f;

#define BATCH 256
#define NPTS  128
#define KNN   16
#define EPSbn 1e-5f

// ----------------------------------------------------------------------------
// KNN (per batch) + out-degree -> dso = rsqrt(max(deg,1))
// block = 128 threads (one per point), grid = 256 batches
// ----------------------------------------------------------------------------
__global__ void knn_kernel(const float* __restrict__ points,   // (B,2,P)
                           const float* __restrict__ mask,     // (B,1,P)
                           int* __restrict__ idx,              // (B*P,K)
                           float* __restrict__ dso)            // (B*P)
{
    const int b = blockIdx.x;
    const int p = threadIdx.x;
    __shared__ float sx[NPTS], sy[NPTS], sm[NPTS];
    __shared__ int   deg[NPTS];
    sx[p]  = points[b * 2 * NPTS + p];
    sy[p]  = points[b * 2 * NPTS + NPTS + p];
    sm[p]  = mask[b * NPTS + p];
    deg[p] = 0;
    __syncthreads();

    const float px = sx[p], py = sy[p];
    float bd[KNN]; int bi[KNN];
#pragma unroll
    for (int k = 0; k < KNN; ++k) { bd[k] = 3.0e38f; bi[k] = 0; }

    // streaming top-K (ascending d2); equal distances keep the earlier index
    for (int j = 0; j < NPTS; ++j) {
        float dx = px - sx[j], dy = py - sy[j];
        float d2 = dx * dx + dy * dy;
        if (sm[j] <= 0.f) d2 = 1.0e9f;
        if (d2 < bd[KNN - 1]) {
            int pos = KNN - 1;
#pragma unroll
            for (int k = KNN - 2; k >= 0; --k) {
                if (d2 < bd[k]) { bd[k + 1] = bd[k]; bi[k + 1] = bi[k]; pos = k; }
            }
            bd[pos] = d2; bi[pos] = j;
        }
    }
#pragma unroll
    for (int k = 0; k < KNN; ++k) {
        idx[(b * NPTS + p) * KNN + k] = bi[k];
        atomicAdd(&deg[bi[k]], 1);
    }
    __syncthreads();
    float d = (float)deg[p];
    dso[b * NPTS + p] = rsqrtf(d > 1.f ? d : 1.f);
}

// ----------------------------------------------------------------------------
// h0[n*34+c] = bn_fts(features[b,c,p])  (transpose + batchnorm)
// ----------------------------------------------------------------------------
__global__ void init_bn_kernel(const float* __restrict__ feats, // (B,34,P)
                               const float* __restrict__ beta,
                               const float* __restrict__ gamma,
                               const float* __restrict__ mean,
                               const float* __restrict__ var,
                               float* __restrict__ h)
{
    int i = blockIdx.x * blockDim.x + threadIdx.x;
    if (i >= BATCH * NPTS * 34) return;
    int c = i % 34;
    int n = i / 34;
    int b = n / NPTS, p = n % NPTS;
    float v = feats[(b * 34 + c) * NPTS + p];
    h[i] = (v - mean[c]) * rsqrtf(var[c] + EPSbn) * gamma[c] + beta[c];
}

// ----------------------------------------------------------------------------
// x = relu(bn(h))  (elementwise, stride C)
// ----------------------------------------------------------------------------
__global__ void bnrelu_kernel(const float* __restrict__ h,
                              const float* __restrict__ beta,
                              const float* __restrict__ gamma,
                              const float* __restrict__ mean,
                              const float* __restrict__ var,
                              float* __restrict__ x, int C)
{
    int i = blockIdx.x * blockDim.x + threadIdx.x;
    if (i >= BATCH * NPTS * C) return;
    int c = i % C;
    float v = h[i];
    v = (v - mean[c]) * rsqrtf(var[c] + EPSbn) * gamma[c] + beta[c];
    x[i] = v > 0.f ? v : 0.f;
}

// ----------------------------------------------------------------------------
// agg[n, c] = 0.25 * sum_k dso[nbr_k] * x[nbr_k, c]  -> f16, padded to Kpad
// block = Kpad threads (one per channel), grid = B*P points
// ----------------------------------------------------------------------------
__global__ void gather_kernel(const float* __restrict__ x,
                              const int* __restrict__ idx,
                              const float* __restrict__ dso,
                              _Float16* __restrict__ aggh,
                              int C, int Kpad)
{
    const int n = blockIdx.x;          // b*P + p
    const int c = threadIdx.x;         // 0..Kpad-1
    const int b = n / NPTS;
    __shared__ int   nb[KNN];
    __shared__ float wgt[KNN];
    if (c < KNN) {
        int g = b * NPTS + idx[n * KNN + c];
        nb[c]  = g;
        wgt[c] = dso[g];
    }
    __syncthreads();
    float acc = 0.f;
    if (c < C) {
#pragma unroll 4
        for (int k = 0; k < KNN; ++k)
            acc += wgt[k] * x[nb[k] * C + c];
        acc *= 0.25f;                  // dsi = 1/sqrt(16)
    }
    aggh[(long)n * Kpad + c] = (_Float16)acc;
}

// ----------------------------------------------------------------------------
// f32 weight (Cin x Cout) -> f16 (Kpad x Cout), zero-padded rows
// ----------------------------------------------------------------------------
__global__ void wconv_kernel(const float* __restrict__ W,
                             _Float16* __restrict__ Wh,
                             int Cin, int Cout, int Kpad)
{
    int i = blockIdx.x * blockDim.x + threadIdx.x;
    if (i >= Kpad * Cout) return;
    int k = i / Cout, co = i % Cout;
    Wh[i] = (k < Cin) ? (_Float16)W[k * Cout + co] : (_Float16)0.f;
}

// ----------------------------------------------------------------------------
// WMMA GEMM: out[M,N] = A[M,Kpad](f16) @ Wh[Kpad,N](f16) + bias (+ xres)
// 8 waves/block; each wave owns a 16x64 output strip (4 accumulator tiles),
// so one A load (2x b128) feeds 4 WMMAs -> 4x less A traffic from L2.
// Block covers 128 rows x 64 cols; 64-col transposed W slice staged in LDS.
// ----------------------------------------------------------------------------
__global__ void __launch_bounds__(256)
gemm_wmma_kernel(const _Float16* __restrict__ A,
                 const _Float16* __restrict__ Wh,
                 const float* __restrict__ bias,
                 const float* __restrict__ xres,
                 float* __restrict__ out,
                 int Kpad, int N, int useRes)
{
    __shared__ __align__(16) _Float16 sW[64 * 256];   // Wt[n][k], 32 KB max
    const int tid  = threadIdx.x;
    const int wave = tid >> 5;
    const int lane = tid & 31;
    const int half = lane >> 4;
    const int l16  = lane & 15;
    const int tileN = blockIdx.y * 64;

    // stage transposed 64-column weight slice: sW[nn*Kpad + kk] = Wh[kk, tileN+nn]
    for (int i = tid; i < 64 * Kpad; i += 256) {
        int nn = i / Kpad, kk = i % Kpad;
        sW[i] = Wh[kk * N + tileN + nn];
    }
    __syncthreads();

    const int  rowBase = (blockIdx.x * 8 + wave) * 16;
    const long aRow    = (long)(rowBase + l16) * Kpad;
    v8f acc0 = {}, acc1 = {}, acc2 = {}, acc3 = {};
    for (int kk = 0; kk < Kpad; kk += 32) {
        // A lane layout (16-bit 16x32): a[e<8] -> K=kk+8*half+e ; a[e>=8] -> K=kk+16+8*half+(e-8)
        v8h a0 = *(const v8h*)(A + aRow + kk + 8 * half);
        v8h a1 = *(const v8h*)(A + aRow + kk + 16 + 8 * half);
        v16h a = __builtin_shufflevector(a0, a1, 0,1,2,3,4,5,6,7,8,9,10,11,12,13,14,15);
        // B lane layout (16-bit 32x16): b[e<8] -> K=kk+8*half+e ; b[e>=8] -> K=kk+16+8*half+(e-8), N=lane%16
#pragma unroll
        for (int nt = 0; nt < 4; ++nt) {
            const _Float16* bp = &sW[(l16 + 16 * nt) * Kpad + kk + 8 * half];
            v8h b0 = *(const v8h*)(bp);
            v8h b1 = *(const v8h*)(bp + 16);
            v16h b = __builtin_shufflevector(b0, b1, 0,1,2,3,4,5,6,7,8,9,10,11,12,13,14,15);
            v8f c = (nt == 0) ? acc0 : (nt == 1) ? acc1 : (nt == 2) ? acc2 : acc3;
            c = __builtin_amdgcn_wmma_f32_16x16x32_f16(false, a, false, b, (short)0, c,
                                                       false, false);
            if (nt == 0) acc0 = c; else if (nt == 1) acc1 = c;
            else if (nt == 2) acc2 = c; else acc3 = c;
        }
    }
    // C/D layout: VGPR r -> row = rowBase + r + 8*half, col = colBase + lane%16
#pragma unroll
    for (int nt = 0; nt < 4; ++nt) {
        const v8f c = (nt == 0) ? acc0 : (nt == 1) ? acc1 : (nt == 2) ? acc2 : acc3;
        const int col = tileN + 16 * nt + l16;
        const float bv = bias[col];
#pragma unroll
        for (int r = 0; r < 8; ++r) {
            int row = rowBase + r + 8 * half;
            float v = c[r] + bv;
            if (useRes) v += xres[(long)row * N + col];
            out[(long)row * N + col] = v;
        }
    }
}

// ----------------------------------------------------------------------------
// masked mean pool over points: hg[b,c] = sum_p m*h / max(sum m, 1)
// ----------------------------------------------------------------------------
__global__ void pool_kernel(const float* __restrict__ h,     // (B*P,256)
                            const float* __restrict__ mask,  // (B,1,P)
                            float* __restrict__ hg)          // (B,256)
{
    int b = blockIdx.x, c = threadIdx.x;
    float s = 0.f, ms = 0.f;
    for (int p = 0; p < NPTS; ++p) {
        float m = mask[b * NPTS + p];
        s  += m * h[(long)(b * NPTS + p) * 256 + c];
        ms += m;
    }
    if (ms < 1.f) ms = 1.f;
    hg[b * 256 + c] = s / ms;
}

// ----------------------------------------------------------------------------
// tiny dense MLP layer: out[b,o] = (relu)(in[b,:] @ W + bias)
// ----------------------------------------------------------------------------
__global__ void mlp_kernel(const float* __restrict__ in, const float* __restrict__ W,
                           const float* __restrict__ bias, float* __restrict__ outp,
                           int din, int dout, int relu)
{
    int i = blockIdx.x * blockDim.x + threadIdx.x;
    if (i >= BATCH * dout) return;
    int b = i / dout, o = i % dout;
    float s = bias[o];
    for (int k = 0; k < din; ++k) s += in[b * din + k] * W[k * dout + o];
    if (relu && s < 0.f) s = 0.f;
    outp[i] = s;
}

// ----------------------------------------------------------------------------
extern "C" void kernel_launch(void* const* d_in, const int* in_sizes, int n_in,
                              void* d_out, int out_size, void* d_ws, size_t ws_size,
                              hipStream_t stream)
{
    (void)out_size; (void)ws_size;
    // ---- resolve input ordering (insertion vs jax-sorted pytree) ----
    int iPoints, iFeatures, iMask, pBase;
    if (in_sizes[0] == BATCH * 2 * NPTS) {           // insertion order
        iPoints = 0; iFeatures = 1; iMask = 3; pBase = 4;
    } else {                                          // sorted: features,lv,mask,params...,points
        iFeatures = 0; iMask = 2; pBase = 3; iPoints = n_in - 1;
    }
    const float* points = (const float*)d_in[iPoints];
    const float* feats  = (const float*)d_in[iFeatures];
    const float* mask   = (const float*)d_in[iMask];
    auto prm = [&](int off) -> const float* { return (const float*)d_in[pBase + off]; };
    // params leaf order (sorted keys): bn_fts{beta,gamma,mean,var}=0..3,
    // bns[i]{beta,gamma,mean,var}=4+4i, convs[i]{W,b}=52+2i, mlp[i](W,b)=76+2i

    // ---- workspace carve-up (256B aligned) ----
    char* ws = (char*)d_ws;
    size_t off = 0;
    auto carve = [&](size_t bytes) -> void* {
        void* p = (void*)(ws + off);
        off += (bytes + 255) & ~(size_t)255;
        return p;
    };
    const int M = BATCH * NPTS;                       // 32768
    int*       idx  = (int*)      carve((size_t)M * KNN * 4);
    float*     dso  = (float*)    carve((size_t)M * 4);
    float*     h    = (float*)    carve((size_t)M * 256 * 4);
    float*     x    = (float*)    carve((size_t)M * 256 * 4);
    _Float16*  agg  = (_Float16*) carve((size_t)M * 256 * 2);
    _Float16*  Wh   = (_Float16*) carve((size_t)256 * 256 * 2);
    float*     hg   = (float*)    carve((size_t)BATCH * 256 * 4);
    float*     y1   = (float*)    carve((size_t)BATCH * 128 * 4);
    float*     y2   = (float*)    carve((size_t)BATCH * 64 * 4);

    // ---- KNN + degree ----
    knn_kernel<<<BATCH, NPTS, 0, stream>>>(points, mask, idx, dso);

    // ---- initial batchnorm (bn_fts) ----
    {
        int total = M * 34;
        init_bn_kernel<<<(total + 255) / 256, 256, 0, stream>>>(
            feats, prm(0), prm(1), prm(2), prm(3), h);
    }

    // ---- 12 graph-conv layers ----
    const int din[12]  = {34, 64, 64, 64, 64, 128, 128, 128, 128, 256, 256, 256};
    const int dout[12] = {64, 64, 64, 64, 128, 128, 128, 128, 256, 256, 256, 256};
    for (int l = 0; l < 12; ++l) {
        const int C = din[l], Co = dout[l];
        const int Kpad = (C + 31) & ~31;
        const float* beta  = prm(4 + 4 * l);
        const float* gamma = prm(5 + 4 * l);
        const float* mean  = prm(6 + 4 * l);
        const float* var   = prm(7 + 4 * l);
        const float* Wc    = prm(52 + 2 * l);
        const float* bc    = prm(53 + 2 * l);

        int tot = M * C;
        bnrelu_kernel<<<(tot + 255) / 256, 256, 0, stream>>>(h, beta, gamma, mean, var, x, C);
        gather_kernel<<<M, Kpad, 0, stream>>>(x, idx, dso, agg, C, Kpad);
        wconv_kernel<<<(Kpad * Co + 255) / 256, 256, 0, stream>>>(Wc, Wh, C, Co, Kpad);
        dim3 grid(M / 128, Co / 64);
        gemm_wmma_kernel<<<grid, 256, 0, stream>>>(agg, Wh, bc, x, h, Kpad, Co,
                                                   (C == Co) ? 1 : 0);
    }

    // ---- pool + MLP head ----
    pool_kernel<<<BATCH, 256, 0, stream>>>(h, mask, hg);
    mlp_kernel<<<(BATCH * 128 + 255) / 256, 256, 0, stream>>>(hg, prm(76), prm(77), y1, 256, 128, 1);
    mlp_kernel<<<(BATCH * 64  + 255) / 256, 256, 0, stream>>>(y1, prm(78), prm(79), y2, 128, 64, 1);
    mlp_kernel<<<(BATCH * 5   + 255) / 256, 256, 0, stream>>>(y2, prm(80), prm(81),
                                                              (float*)d_out, 64, 5, 0);
}
